// SparseMoE_19387482374733
// MI455X (gfx1250) — matrix operand order repaired and checked by reading, not compiled, not run on hardware
//
#include <hip/hip_runtime.h>
#include <hip/hip_bf16.h>

#define D_EMB 768
#define N_EXP 8
#define H_DIM 3072
#define TSTRIDE 72          // LDS row stride in halves for a 64-half row (+8 pad)

typedef __attribute__((ext_vector_type(16))) __bf16        v16bf;
typedef __attribute__((ext_vector_type(8)))  float         v8f;
typedef __attribute__((ext_vector_type(4)))  unsigned int  v4u;
typedef __attribute__((ext_vector_type(8)))  unsigned int  v8u;

static __device__ __forceinline__ unsigned short f32_to_bf16_rne(float f) {
    unsigned int u = __builtin_bit_cast(unsigned int, f);
    u += 0x7FFFu + ((u >> 16) & 1u);
    return (unsigned short)(u >> 16);
}

// ---------------------------------------------------------------- utilities
__global__ __launch_bounds__(256) void moe_zero_f32(float* __restrict__ p, long n) {
    long i = (long)blockIdx.x * blockDim.x + threadIdx.x;
    long s = (long)gridDim.x * blockDim.x;
    for (; i < n; i += s) p[i] = 0.0f;
}

__global__ __launch_bounds__(256) void moe_cvt_bf16(const float* __restrict__ x,
                                                    unsigned short* __restrict__ y, long n) {
    long i = (long)blockIdx.x * blockDim.x + threadIdx.x;
    long s = (long)gridDim.x * blockDim.x;
    for (; i < n; i += s) y[i] = f32_to_bf16_rne(x[i]);
}

// Transpose + convert one weight matrix: W[K][N] f32 -> WT[N][K] bf16 (once per expert)
__global__ __launch_bounds__(256) void moe_wt_cvt(const float* __restrict__ W,
                                                  unsigned short* __restrict__ WT,
                                                  int Kd, int Nd) {
    __shared__ unsigned short t[64 * 65];
    const int nb = blockIdx.x * 64, kb = blockIdx.y * 64;
    const int tid = threadIdx.x;
#pragma unroll
    for (int i = 0; i < 16; ++i) {
        int idx = tid + 256 * i;
        int r = idx >> 6, c = idx & 63;            // r = k-local, c = n-local
        t[c * 65 + r] = f32_to_bf16_rne(W[(size_t)(kb + r) * Nd + nb + c]);
    }
    __syncthreads();
#pragma unroll
    for (int i = 0; i < 16; ++i) {
        int idx = tid + 256 * i;
        int r = idx >> 6, c = idx & 63;            // r = n-local, c = k-local
        WT[(size_t)(nb + r) * Kd + kb + c] = t[r * 65 + c];
    }
}

// ---------------------------------------------------------------- router
__global__ __launch_bounds__(256) void moe_router(const float* __restrict__ x,
                                                  const float* __restrict__ Wg,
                                                  const float* __restrict__ bg,
                                                  float* __restrict__ gates, int T) {
    const int lane = threadIdx.x & 31;
    const int tok  = blockIdx.x * 8 + (threadIdx.x >> 5);
    if (tok >= T) return;

    float s[N_EXP];
#pragma unroll
    for (int e = 0; e < N_EXP; ++e) s[e] = 0.0f;

    const float* xr = x + (size_t)tok * D_EMB;
    for (int d = lane; d < D_EMB; d += 32) {
        float xv = xr[d];
        const float* wr = Wg + (size_t)d * N_EXP;
#pragma unroll
        for (int e = 0; e < N_EXP; ++e) s[e] += xv * wr[e];
    }
#pragma unroll
    for (int off = 16; off > 0; off >>= 1) {
#pragma unroll
        for (int e = 0; e < N_EXP; ++e) s[e] += __shfl_xor(s[e], off, 32);
    }
    if (lane == 0) {
#pragma unroll
        for (int e = 0; e < N_EXP; ++e) s[e] += bg[e];
        int i1 = 0; float v1 = s[0];
#pragma unroll
        for (int e = 1; e < N_EXP; ++e) if (s[e] > v1) { v1 = s[e]; i1 = e; }
        int i2 = -1; float v2 = -3.4e38f;
#pragma unroll
        for (int e = 0; e < N_EXP; ++e) if (e != i1 && s[e] > v2) { v2 = s[e]; i2 = e; }
        float p1 = 1.0f / (1.0f + __expf(v2 - v1));
        float p2 = 1.0f - p1;
        float* gr = gates + (size_t)tok * N_EXP;
#pragma unroll
        for (int e = 0; e < N_EXP; ++e) gr[e] = (e == i1) ? p1 : ((e == i2) ? p2 : 0.0f);
    }
}

// ---------------------------------------------------------------- WMMA GEMM core
// A[M][K] bf16, BT[N][K] bf16 (row-major, K contiguous).
// Block = 256 threads (8 wave32) arranged 4(M) x 2(N); each wave owns a
// 32x64 sub-tile (2x4 WMMA tiles). 128x128 block tile, K-step 64,
// double-buffered LDS filled via global_load_async_to_lds_b128 (ASYNCcnt).

static __device__ __forceinline__ v16bf frag_from_lds(const unsigned short* base,
                                                      int off1) {
    v4u a0 = *(const v4u*)base;
    v4u a1 = *(const v4u*)(base + off1);
    v8u u;
    u[0] = a0[0]; u[1] = a0[1]; u[2] = a0[2]; u[3] = a0[3];
    u[4] = a1[0]; u[5] = a1[1]; u[6] = a1[2]; u[7] = a1[3];
    return __builtin_bit_cast(v16bf, u);
}

static __device__ __forceinline__ void async_tile16(unsigned short* ldsp,
                                                    const unsigned short* gp) {
    unsigned loff = (unsigned)(uintptr_t)ldsp;       // low 32 bits == LDS offset
    asm volatile("global_load_async_to_lds_b128 %0, %1, off"
                 :: "v"(loff), "v"(gp) : "memory");
}

// stage one 128x64-half tile of A and of BT into LDS buffers (8 async b128/thread)
static __device__ __forceinline__ void stage_async(unsigned short* Albuf,
                                                   unsigned short* Blbuf,
                                                   const unsigned short* __restrict__ A,
                                                   const unsigned short* __restrict__ BT,
                                                   int ld, int mbase, int nbase,
                                                   int k0, int tid) {
#pragma unroll
    for (int i = 0; i < 4; ++i) {
        int ch  = tid + 256 * i;      // 1024 chunks of 8 halves
        int row = ch >> 3;
        int cg  = ch & 7;
        async_tile16(Albuf + row * TSTRIDE + cg * 8,
                     A + (size_t)(mbase + row) * ld + k0 + cg * 8);
        async_tile16(Blbuf + row * TSTRIDE + cg * 8,
                     BT + (size_t)(nbase + row) * ld + k0 + cg * 8);
    }
}

// wave computes 2(M) x 4(N) tiles; all 6 fragments preloaded into distinct
// registers so the 8 WMMAs issue back-to-back behind partial dscnt waits.
static __device__ __forceinline__ void compute64(const unsigned short* Ac,
                                                 const unsigned short* Bc,
                                                 v8f acc[8], int mg, int ng,
                                                 int l15, int lhi) {
#pragma unroll
    for (int kk = 0; kk < 64; kk += 32) {
        v16bf af[2];
#pragma unroll
        for (int mt = 0; mt < 2; ++mt)
            af[mt] = frag_from_lds(
                Ac + (mg * 32 + mt * 16 + l15) * TSTRIDE + kk + lhi * 8, 16);
        v16bf bfv[4];
#pragma unroll
        for (int nt = 0; nt < 4; ++nt)
            bfv[nt] = frag_from_lds(
                Bc + (ng * 64 + nt * 16 + l15) * TSTRIDE + kk + lhi * 8, 16);
#pragma unroll
        for (int mt = 0; mt < 2; ++mt)
#pragma unroll
            for (int nt = 0; nt < 4; ++nt)
                acc[mt * 4 + nt] = __builtin_amdgcn_wmma_f32_16x16x32_bf16(
                    false, af[mt], false, bfv[nt], (short)0, acc[mt * 4 + nt],
                    false, false);
    }
}

// branch-free hot loop + peeled final K-iteration
#define GEMM_MAINLOOP(KTOT)                                                         \
    __shared__ __align__(32) unsigned short Al[2 * 128 * TSTRIDE];                  \
    __shared__ __align__(32) unsigned short Bl[2 * 128 * TSTRIDE];                  \
    const int tid  = threadIdx.x;                                                   \
    const int lane = tid & 31;                                                      \
    const int wv   = tid >> 5;                                                      \
    const int mg   = wv >> 1;                                                       \
    const int ng   = wv & 1;                                                        \
    const int l15  = lane & 15;                                                     \
    const int lhi  = lane >> 4;                                                     \
    const int nbase = blockIdx.x * 128;                                             \
    const int mbase = blockIdx.y * 128;                                             \
    v8f acc[8];                                                                     \
    _Pragma("unroll")                                                               \
    for (int i = 0; i < 8; ++i) acc[i] = (v8f){0,0,0,0,0,0,0,0};                    \
    stage_async(Al, Bl, A, BT, KTOT, mbase, nbase, 0, tid);                         \
    for (int k0 = 0; k0 < (KTOT) - 64; k0 += 64) {                                  \
        const int cur = (k0 >> 6) & 1;                                              \
        const unsigned short* Ac = Al + cur * 128 * TSTRIDE;                        \
        const unsigned short* Bc = Bl + cur * 128 * TSTRIDE;                        \
        stage_async(Al + (cur ^ 1) * 128 * TSTRIDE,                                 \
                    Bl + (cur ^ 1) * 128 * TSTRIDE,                                 \
                    A, BT, KTOT, mbase, nbase, k0 + 64, tid);                       \
        asm volatile("s_wait_asynccnt 0x8" ::: "memory");                           \
        __syncthreads();                                                            \
        compute64(Ac, Bc, acc, mg, ng, l15, lhi);                                   \
        __syncthreads();                                                            \
    }                                                                               \
    {                                                                               \
        const int cur = (((KTOT) - 64) >> 6) & 1;                                   \
        const unsigned short* Ac = Al + cur * 128 * TSTRIDE;                        \
        const unsigned short* Bc = Bl + cur * 128 * TSTRIDE;                        \
        asm volatile("s_wait_asynccnt 0x0" ::: "memory");                           \
        __syncthreads();                                                            \
        compute64(Ac, Bc, acc, mg, ng, l15, lhi);                                   \
    }

// FC1:  H = relu(x[M,768] @ W1 + b1) -> bf16      (BT = W1^T as [3072][768] bf16)
__global__ __launch_bounds__(256) void moe_fc1(const unsigned short* __restrict__ A,
                                               const unsigned short* __restrict__ BT,
                                               const float* __restrict__ bias,
                                               unsigned short* __restrict__ Hout,
                                               int M) {
    (void)M;
    GEMM_MAINLOOP(D_EMB)
#pragma unroll
    for (int mt = 0; mt < 2; ++mt)
#pragma unroll
        for (int nt = 0; nt < 4; ++nt) {
            int n = nbase + ng * 64 + nt * 16 + l15;
            float b = bias[n];
#pragma unroll
            for (int r = 0; r < 8; ++r) {
                int m = mbase + mg * 32 + mt * 16 + lhi * 8 + r;
                float v = acc[mt * 4 + nt][r] + b;
                v = v > 0.0f ? v : 0.0f;
                Hout[(size_t)m * H_DIM + n] = f32_to_bf16_rne(v);
            }
        }
}

// FC2:  out[tok,:] += gate * (H[M,3072] @ W2 + b2)   (BT = W2^T as [768][3072] bf16)
__global__ __launch_bounds__(256) void moe_fc2(const unsigned short* __restrict__ A,
                                               const unsigned short* __restrict__ BT,
                                               const float* __restrict__ bias,
                                               const float* __restrict__ gates,
                                               float* __restrict__ out,
                                               int tok0, int expert, int M) {
    (void)M;
    GEMM_MAINLOOP(H_DIM)
    float g[2][8];
#pragma unroll
    for (int mt = 0; mt < 2; ++mt)
#pragma unroll
        for (int r = 0; r < 8; ++r) {
            int m = mbase + mg * 32 + mt * 16 + lhi * 8 + r;
            g[mt][r] = gates[(size_t)(tok0 + m) * N_EXP + expert];
        }
#pragma unroll
    for (int mt = 0; mt < 2; ++mt)
#pragma unroll
        for (int nt = 0; nt < 4; ++nt) {
            int n = nbase + ng * 64 + nt * 16 + l15;
            float b = bias[n];
#pragma unroll
            for (int r = 0; r < 8; ++r) {
                int m = mbase + mg * 32 + mt * 16 + lhi * 8 + r;
                float* o = out + (size_t)(tok0 + m) * D_EMB + n;
                *o += g[mt][r] * (acc[mt * 4 + nt][r] + b);
            }
        }
}

// ---------------------------------------------------------------- launcher
extern "C" void kernel_launch(void* const* d_in, const int* in_sizes, int n_in,
                              void* d_out, int out_size, void* d_ws, size_t ws_size,
                              hipStream_t stream) {
    const float* x  = (const float*)d_in[0];
    const float* Wg = (const float*)d_in[1];
    const float* bg = (const float*)d_in[2];
    const float* W1 = (const float*)d_in[3];
    const float* b1 = (const float*)d_in[4];
    const float* W2 = (const float*)d_in[5];
    const float* b2 = (const float*)d_in[6];
    float* out = (float*)d_out;

    const int T = in_sizes[0] / D_EMB;          // 16384

    // ws layout: x_bf16 | gates | W1e^T bf16 | W2e^T bf16 | h_chunk
    size_t off_x  = 0;
    size_t sz_x   = (size_t)T * D_EMB * sizeof(unsigned short);
    size_t off_g  = (off_x + sz_x + 255) & ~(size_t)255;
    size_t sz_g   = (size_t)T * N_EXP * sizeof(float);
    size_t off_w1 = (off_g + sz_g + 255) & ~(size_t)255;
    size_t sz_w   = (size_t)D_EMB * H_DIM * sizeof(unsigned short);
    size_t off_w2 = (off_w1 + sz_w + 255) & ~(size_t)255;
    size_t off_h  = (off_w2 + sz_w + 255) & ~(size_t)255;

    int Tc = T;                                  // prefer one chunk (h = T*H bf16)
    while (Tc > 128 && off_h + (size_t)Tc * H_DIM * sizeof(unsigned short) > ws_size)
        Tc >>= 1;

    unsigned short* xbf   = (unsigned short*)((char*)d_ws + off_x);
    float*          gates = (float*)((char*)d_ws + off_g);
    unsigned short* w1t   = (unsigned short*)((char*)d_ws + off_w1);
    unsigned short* w2t   = (unsigned short*)((char*)d_ws + off_w2);
    unsigned short* hbuf  = (unsigned short*)((char*)d_ws + off_h);

    const long nTD = (long)T * D_EMB;
    moe_zero_f32<<<2048, 256, 0, stream>>>(out, nTD);
    moe_cvt_bf16<<<2048, 256, 0, stream>>>(x, xbf, nTD);
    moe_router<<<(T + 7) / 8, 256, 0, stream>>>(x, Wg, bg, gates, T);

    for (int e = 0; e < N_EXP; ++e) {
        moe_wt_cvt<<<dim3(H_DIM / 64, D_EMB / 64), 256, 0, stream>>>(
            W1 + (size_t)e * D_EMB * H_DIM, w1t, D_EMB, H_DIM);
        moe_wt_cvt<<<dim3(D_EMB / 64, H_DIM / 64), 256, 0, stream>>>(
            W2 + (size_t)e * H_DIM * D_EMB, w2t, H_DIM, D_EMB);

        for (int c0 = 0; c0 < T; c0 += Tc) {
            int Mc = (T - c0 < Tc) ? (T - c0) : Tc;
            dim3 g1(H_DIM / 128, Mc / 128);
            moe_fc1<<<g1, 256, 0, stream>>>(xbf + (size_t)c0 * D_EMB, w1t,
                                            b1 + (size_t)e * H_DIM, hbuf, Mc);
            dim3 g2(D_EMB / 128, Mc / 128);
            moe_fc2<<<g2, 256, 0, stream>>>(hbuf, w2t,
                                            b2 + (size_t)e * D_EMB,
                                            gates, out, c0, e, Mc);
        }
    }
}
